// PerspectiveNetwork_42408507081247
// MI455X (gfx1250) — matrix operand order repaired and checked by reading, not compiled, not run on hardware
//
#include <hip/hip_runtime.h>
#include <math.h>

typedef __attribute__((ext_vector_type(16))) _Float16 v16h;
typedef __attribute__((ext_vector_type(8)))  _Float16 v8h;
typedef __attribute__((ext_vector_type(8)))  float    v8f;

#define F_IN  768
#define F_OUT 512

__device__ __forceinline__ v16h cvt16(float4 a, float4 b, float4 c, float4 d) {
    v16h r;
    r[0]  = (_Float16)a.x; r[1]  = (_Float16)a.y; r[2]  = (_Float16)a.z; r[3]  = (_Float16)a.w;
    r[4]  = (_Float16)b.x; r[5]  = (_Float16)b.y; r[6]  = (_Float16)b.z; r[7]  = (_Float16)b.w;
    r[8]  = (_Float16)c.x; r[9]  = (_Float16)c.y; r[10] = (_Float16)c.z; r[11] = (_Float16)c.w;
    r[12] = (_Float16)d.x; r[13] = (_Float16)d.y; r[14] = (_Float16)d.z; r[15] = (_Float16)d.w;
    return r;
}

__device__ __forceinline__ float4 ld4(const float* p) { return *(const float4*)p; }

__device__ __forceinline__ v16h ld16h(const _Float16* p) {
    v8h lo = *(const v8h*)(p);
    v8h hi = *(const v8h*)(p + 8);
    v16h r;
    #pragma unroll
    for (int i = 0; i < 8; ++i) { r[i] = lo[i]; r[i + 8] = hi[i]; }
    return r;
}

// Pre-pass: convert W1 (fp32, 512x768 row-major) to f16 in workspace, same layout.
__global__ __launch_bounds__(256)
void convert_w1_kernel(const float* __restrict__ W1, _Float16* __restrict__ W1h) {
    const int i = (blockIdx.x * 256 + threadIdx.x) * 8;   // 192 blocks cover 393216
    float4 a = ld4(W1 + i);
    float4 b = ld4(W1 + i + 4);
    v8h r;
    r[0] = (_Float16)a.x; r[1] = (_Float16)a.y; r[2] = (_Float16)a.z; r[3] = (_Float16)a.w;
    r[4] = (_Float16)b.x; r[5] = (_Float16)b.y; r[6] = (_Float16)b.z; r[7] = (_Float16)b.w;
    *(v8h*)(W1h + i) = r;
}

// Fused: [32 rows/WG] x (stm@W1.T, nstm@W1.T) -> +b1 -> clip01 -> dot W2 -> sigmoid.
// 8 waves: waveM (0..1) picks 16-row slab, waveN (0..3) picks 128-col stripe.
// Each wave: 8 N-tiles x 2 perspectives = 16 f32 16x16 accumulators.
template <bool F16W>
__global__ __launch_bounds__(256, 2)
void nnue_fused_kernel(const float* __restrict__ stm,  const float* __restrict__ nstm,
                       const float* __restrict__ W1f,  const _Float16* __restrict__ W1h,
                       const float* __restrict__ b1,
                       const float* __restrict__ W2,   const float* __restrict__ b2,
                       float* __restrict__ out)
{
    __shared__ float srow[32];

    const int tid   = threadIdx.x;
    const int lane  = tid & 31;
    const int wid   = tid >> 5;
    const int half  = lane >> 4;     // 0: lanes 0-15, 1: lanes 16-31
    const int l15   = lane & 15;
    const int waveM = wid >> 2;      // 0..1
    const int waveN = wid & 3;       // 0..3

    const int rowBase = blockIdx.x * 32 + waveM * 16;   // batch rows of this wave
    const int colBase = waveN * 128;                    // output features of this wave

    if (tid < 32) srow[tid] = 0.0f;
    __syncthreads();

    // A-fragment per-lane base: row (rowBase+l15), 16 contiguous K at half*16 per k-tile.
    const float* aS = stm  + (rowBase + l15) * F_IN + half * 16;
    const float* aN = nstm + (rowBase + l15) * F_IN + half * 16;
    // B-fragment per-lane base: W1 row (col n = colBase + nt*16 + l15), 16 contiguous K at half*16.
    const float*    bBf = W1f + (colBase + l15) * F_IN + half * 16;
    const _Float16* bBh = W1h + (colBase + l15) * F_IN + half * 16;

    v8f accS[8], accN[8];
    #pragma unroll
    for (int nt = 0; nt < 8; ++nt) { v8f z = {}; accS[nt] = z; accN[nt] = z; }

    for (int kt = 0; kt < F_IN; kt += 32) {
        // A fragments (16x32 f16): lane<16 holds K 0..15, lane>=16 holds K 16..31
        v16h aSf = cvt16(ld4(aS + kt),      ld4(aS + kt + 4),
                         ld4(aS + kt + 8),  ld4(aS + kt + 12));
        v16h aNf = cvt16(ld4(aN + kt),      ld4(aN + kt + 4),
                         ld4(aN + kt + 8),  ld4(aN + kt + 12));
        #pragma unroll
        for (int nt = 0; nt < 8; ++nt) {
            // B fragment (32x16 f16): lane<16 => K 0..15 of column n, lane>=16 => K 16..31.
            v16h bf;
            if constexpr (F16W) {
                bf = ld16h(bBh + kt + nt * (16 * F_IN));
            } else {
                const float* bp = bBf + kt + nt * (16 * F_IN);
                bf = cvt16(ld4(bp), ld4(bp + 4), ld4(bp + 8), ld4(bp + 12));
            }
            accS[nt] = __builtin_amdgcn_wmma_f32_16x16x32_f16(
                false, aSf, false, bf, (short)0, accS[nt], false, false);
            accN[nt] = __builtin_amdgcn_wmma_f32_16x16x32_f16(
                false, aNf, false, bf, (short)0, accN[nt], false, false);
        }
    }

    // Per-lane epilogue constants: bias + W2 weights for this lane's output column.
    float b1v[8], w2s[8], w2n[8];
    #pragma unroll
    for (int nt = 0; nt < 8; ++nt) {
        const int c = colBase + nt * 16 + l15;
        b1v[nt] = b1[c];
        w2s[nt] = W2[c];
        w2n[nt] = W2[F_OUT + c];
    }

    // C/D layout: VGPR v, lanes 0-15 -> (row v, col l15); lanes 16-31 -> (row v+8, col l15).
    #pragma unroll
    for (int v = 0; v < 8; ++v) {
        float s = 0.0f;
        #pragma unroll
        for (int nt = 0; nt < 8; ++nt) {
            float hS = accS[nt][v] + b1v[nt];
            float hN = accN[nt][v] + b1v[nt];
            hS = fminf(fmaxf(hS, 0.0f), 1.0f);
            hN = fminf(fmaxf(hN, 0.0f), 1.0f);
            s += hS * w2s[nt] + hN * w2n[nt];
        }
        // reduce across the 16 lanes holding the same output row
        #pragma unroll
        for (int m = 1; m < 16; m <<= 1) s += __shfl_xor(s, m, 16);
        if (l15 == 0) atomicAdd(&srow[waveM * 16 + half * 8 + v], s);  // ds_add_f32
    }
    __syncthreads();

    if (tid < 32) {
        const float x = srow[tid] + b2[0];
        out[blockIdx.x * 32 + tid] = 1.0f / (1.0f + __expf(-x));
    }
}

extern "C" void kernel_launch(void* const* d_in, const int* in_sizes, int n_in,
                              void* d_out, int out_size, void* d_ws, size_t ws_size,
                              hipStream_t stream) {
    const float* stm  = (const float*)d_in[0];
    const float* nstm = (const float*)d_in[1];
    const float* W1   = (const float*)d_in[2];
    const float* b1   = (const float*)d_in[3];
    const float* W2   = (const float*)d_in[4];
    const float* b2   = (const float*)d_in[5];
    float* out = (float*)d_out;

    const int B = in_sizes[0] / F_IN;     // 16384
    const size_t w1h_bytes = (size_t)F_OUT * F_IN * sizeof(_Float16);  // 768 KB

    dim3 grid(B / 32), block(256);

    if (ws_size >= w1h_bytes) {
        _Float16* W1h = (_Float16*)d_ws;
        convert_w1_kernel<<<dim3((F_OUT * F_IN) / (256 * 8)), block, 0, stream>>>(W1, W1h);
        nnue_fused_kernel<true><<<grid, block, 0, stream>>>(
            stm, nstm, W1, W1h, b1, W2, b2, out);
    } else {
        nnue_fused_kernel<false><<<grid, block, 0, stream>>>(
            stm, nstm, W1, (const _Float16*)nullptr, b1, W2, b2, out);
    }
}
